// DiagonalBiLSTM_40707700032184
// MI455X (gfx1250) — compile-verified, matrix operand order checked by reading, not run on hardware
//
#include <hip/hip_runtime.h>
#include <hip/hip_bf16.h>

// DiagonalBiLSTM (PixelRNN) for MI455X / gfx1250.
// Latency-bound diagonal scan -> 1 WG per (batch,direction), weights+state in
// LDS, bf16 WMMA (v_wmma_f32_16x16x32_bf16) with fp32 accumulation, LSTM cell
// state and gating held entirely in registers (gate-aligned output tiling).

constexpr int Bz = 32, Mz = 64, Nz = 64, Fz = 64, F2z = 128, N2z = 127;

typedef __attribute__((ext_vector_type(16))) __bf16 v16bf;
typedef __attribute__((ext_vector_type(8)))  float  v8f;

union AF { v16bf v; uint4 q[2]; };

__device__ __forceinline__ v8f wmma_bf16(v16bf a, v16bf b, v8f c) {
  // 8 args: (neg_a, A, neg_b, B, c_mod, C, reuse_a, reuse_b)
  return __builtin_amdgcn_wmma_f32_16x16x32_bf16(false, a, false, b, (short)0, c, false, false);
}

__device__ __forceinline__ float sigmoidf_(float x) { return 1.0f / (1.0f + __expf(-x)); }

// ---------------------------------------------------------------------------
// Weight swizzle: fp32 row-major (K x N) -> bf16 WMMA-B fragment order.
// Fragment (kt,nt) of 32x16, lane-major, 16 contiguous bf16 per lane.
// 16-bit operand K map per element e, half h: k = ((e>>3)<<4)+(((e>>1)&3)<<1)+(e&1)+8h
// mode 0: src[k*N+n].  mode 1 (is_w, shape (4,K,64)): src[(n>>6)*K*64 + k*64 + (n&63)]
// ---------------------------------------------------------------------------
__global__ void __launch_bounds__(256)
swizzle_kernel(const float* __restrict__ src, __bf16* __restrict__ dst,
               int K, int N, int mode) {
  int i = blockIdx.x * 256 + threadIdx.x;
  if (i >= K * N) return;
  int e  = i & 15;
  int ln = (i >> 4) & 31;
  int fr = i >> 9;
  int ntiles = N >> 4;
  int nt = fr % ntiles, kt = fr / ntiles;
  int hf = ln >> 4;
  int k = kt * 32 + ((e >> 3) << 4) + (((e >> 1) & 3) << 1) + (e & 1) + hf * 8;
  int n = nt * 16 + (ln & 15);
  float v = (mode == 0) ? src[k * N + n]
                        : src[(n >> 6) * (K * 64) + k * 64 + (n & 63)];
  dst[i] = (__bf16)v;
}

// ---------------------------------------------------------------------------
// Strictly-causal masked 7x7 conv, 1 -> 128 channels. One block per pixel.
// ---------------------------------------------------------------------------
__global__ void __launch_bounds__(128)
inconv_kernel(const float* __restrict__ im, const float* __restrict__ w,
              const float* __restrict__ bconv,
              float* __restrict__ xf32, __bf16* __restrict__ xbf) {
  __shared__ float patch[28];
  int pix = blockIdx.x;
  int n = pix & 63, m = (pix >> 6) & 63, bb = pix >> 12;
  int c = threadIdx.x;
  if (c < 28) {
    int dy = c / 7, dx = c % 7;
    int y = m + dy - 3, x = n + dx - 3;
    patch[c] = (y >= 0 && y < Mz && x >= 0 && x < Nz)
             ? im[((size_t)bb * Mz + y) * Nz + x] : 0.0f;
  }
  __syncthreads();
  float acc = bconv[c];
  #pragma unroll
  for (int dy = 0; dy < 3; ++dy)
    #pragma unroll
    for (int dx = 0; dx < 7; ++dx)
      acc += patch[dy * 7 + dx] * w[(dy * 7 + dx) * F2z + c];
  #pragma unroll
  for (int dx = 0; dx < 3; ++dx)          // center row, left of center only
    acc += patch[21 + dx] * w[(21 + dx) * F2z + c];
  size_t off = (size_t)pix * F2z + c;
  xf32[off] = acc;
  xbf[off]  = (__bf16)acc;
}

// ---------------------------------------------------------------------------
// Diagonal LSTM scan. grid.x = 64: blocks 0..31 = left dir, 32..63 = right.
// Wave w owns rows mt = w>>1 and f-columns (w&1)*32..+31 (two 16-col tiles).
// Gate-aligned tiling: gate g of column f is output tile nt = (f>>4) + 4g, so
// fz/iz/oz/gz for (m,f) sit in the same lane/element of 4 accumulators ->
// gating + cell state fully in registers; h -> LDS bf16 for next step.
// ---------------------------------------------------------------------------
__global__ void __launch_bounds__(256)
scan_kernel(const __bf16* __restrict__ xbf, const float* __restrict__ xf32,
            const __bf16* __restrict__ wL, const __bf16* __restrict__ wR,
            const float* __restrict__ isbL, const float* __restrict__ ssbL,
            const float* __restrict__ outbL, const float* __restrict__ h0L,
            const float* __restrict__ isbR, const float* __restrict__ ssbR,
            const float* __restrict__ outbR, const float* __restrict__ h0R,
            float* __restrict__ leftU, float* __restrict__ rightU) {
  extern __shared__ char smem[];
  __bf16* sWis  = (__bf16*)smem;            // 128x256 = 32768 el (swizzled)
  __bf16* sWss  = sWis + 32768;             // 128x256 (rows 0..63 ss_w[0], 64..127 ss_w[1])
  __bf16* sWout = sWss + 32768;             // 64x128 = 8192 el
  __bf16* sHbf  = sWout + 8192;             // 64x64 hidden (bf16)

  const int tid  = threadIdx.x;
  const int wave = tid >> 5;
  const int lane = tid & 31;
  const int hf   = lane >> 4;
  const int l15  = lane & 15;
  const int b    = blockIdx.x & (Bz - 1);
  const int dir  = blockIdx.x >> 5;

  const int mt = wave >> 1;                 // row-tile owned by this wave
  const int fh = wave & 1;                  // f-column half (0: f<32, 1: f>=32)
  const int m  = mt * 16 + l15;             // A-matrix row for this lane

  const __bf16* wdir = dir ? wR : wL;
  const float* isb  = dir ? isbR  : isbL;
  const float* ssb  = dir ? ssbR  : ssbL;
  const float* outb = dir ? outbR : outbL;
  const float* h0   = dir ? h0R   : h0L;
  float* outU       = dir ? rightU : leftU;

  // loop-invariant biases -> registers
  float bz[2][4];                           // [local f-tile][gate]
  #pragma unroll
  for (int fl = 0; fl < 2; ++fl) {
    int f = (fh * 2 + fl) * 16 + l15;
    #pragma unroll
    for (int g = 0; g < 4; ++g) bz[fl][g] = isb[g * 64 + f] + ssb[g * 64 + f];
  }
  float ob[4];
  #pragma unroll
  for (int t = 0; t < 4; ++t) ob[t] = outb[fh * 64 + t * 16 + l15];

  { // weights -> LDS (wis|wss|wout contiguous, 73728 bf16 = 9216 uint4)
    const uint4* src = (const uint4*)wdir;
    uint4* dst = (uint4*)smem;
    for (int i = tid; i < 9216; i += 256) dst[i] = src[i];
  }
  for (int i = tid; i < 64 * 64; i += 256)  // h init (h0 broadcast over m)
    sHbf[i] = (__bf16)h0[i & 63];
  v8f cst[2] = {{}, {}};                    // cell state, registers
  __syncthreads();

  #pragma unroll 1
  for (int j = 0; j < N2z; ++j) {
    // ---- A-fragments (shared across all 8 output tiles of this wave)
    int colX = (dir == 0) ? (j - m) : (N2z - 1 - j - m);
    bool vx = (colX >= 0) && (colX < Nz);
    const __bf16* xrow =
        xbf + (((size_t)b * Mz + m) * Nz + (vx ? colX : 0)) * F2z;
    AF ax[4], ah[4];
    #pragma unroll
    for (int kt = 0; kt < 4; ++kt) {
      int k0 = kt * 32 + hf * 8;
      if (vx) {
        ax[kt].q[0] = *(const uint4*)(xrow + k0);
        ax[kt].q[1] = *(const uint4*)(xrow + k0 + 16);
      } else {
        ax[kt].q[0] = make_uint4(0, 0, 0, 0);
        ax[kt].q[1] = make_uint4(0, 0, 0, 0);
      }
      #pragma unroll
      for (int ch = 0; ch < 2; ++ch) {      // Hcat = [h[m-1] | h[m]]
        int ks  = k0 + ch * 16;
        int row = (ks < 64) ? (m - 1) : m;
        int kk  = (ks < 64) ? ks : (ks - 64);
        ah[kt].q[ch] = (row >= 0) ? *(const uint4*)(sHbf + row * 64 + kk)
                                  : make_uint4(0, 0, 0, 0);
      }
    }
    __syncthreads();  // all h_{j-1} reads done before gating overwrites sHbf

    // ---- gates (WMMA) + LSTM cell update (registers)
    #pragma unroll
    for (int fl = 0; fl < 2; ++fl) {
      int ft = fh * 2 + fl;                 // f-tile index 0..3
      v8f acc[4] = {{}, {}, {}, {}};
      #pragma unroll
      for (int g = 0; g < 4; ++g) {
        int nt = ft + g * 4;                // gate-aligned output tile
        #pragma unroll
        for (int kt = 0; kt < 4; ++kt)      // i2s : K = 128
          acc[g] = wmma_bf16(ax[kt].v,
              *(const v16bf*)(sWis + (kt * 16 + nt) * 512 + lane * 16), acc[g]);
        #pragma unroll
        for (int kt = 0; kt < 4; ++kt)      // s2s : K = 128
          acc[g] = wmma_bf16(ah[kt].v,
              *(const v16bf*)(sWss + (kt * 16 + nt) * 512 + lane * 16), acc[g]);
      }
      #pragma unroll
      for (int v = 0; v < 8; ++v) {         // D layout: m = mt*16 + 8*hf + v
        float fz = acc[0][v] + bz[fl][0];
        float iz = acc[1][v] + bz[fl][1];
        float oz = acc[2][v] + bz[fl][2];
        float gz = acc[3][v] + bz[fl][3];
        float cn = sigmoidf_(fz) * cst[fl][v] + sigmoidf_(iz) * tanhf(gz);
        float hn = sigmoidf_(oz) * tanhf(cn);
        cst[fl][v] = cn;
        sHbf[(mt * 16 + hf * 8 + v) * 64 + ft * 16 + l15] = (__bf16)hn;
      }
    }
    __syncthreads();  // h_j complete before Phase C / next-step reads

    // ---- Phase C: Y = h@Wout + out_b + residual, unskewed store
    AF a0, a1;                              // hoisted A-frags (K = 64)
    a0.q[0] = *(const uint4*)(sHbf + m * 64 + hf * 8);
    a0.q[1] = *(const uint4*)(sHbf + m * 64 + hf * 8 + 16);
    a1.q[0] = *(const uint4*)(sHbf + m * 64 + 32 + hf * 8);
    a1.q[1] = *(const uint4*)(sHbf + m * 64 + 32 + hf * 8 + 16);
    int jj = (dir == 0) ? j : (N2z - 1 - j);
    #pragma unroll
    for (int t = 0; t < 4; ++t) {
      int nt = fh * 4 + t;
      v8f acc = {};
      acc = wmma_bf16(a0.v, *(const v16bf*)(sWout + nt * 512 + lane * 16), acc);
      acc = wmma_bf16(a1.v, *(const v16bf*)(sWout + (8 + nt) * 512 + lane * 16), acc);
      #pragma unroll
      for (int v = 0; v < 8; ++v) {
        int mo   = mt * 16 + hf * 8 + v;    // scan row
        int o    = nt * 16 + l15;           // channel
        int rowT = (dir == 0) ? mo : (mo + 1);  // right dir shifted down by 1
        int colT = jj - rowT;               // unskew
        if (rowT < Mz && colT >= 0 && colT < Nz) {
          int cX = jj - mo;                 // residual uses unshifted row
          float res = (cX >= 0 && cX < Nz)
              ? xf32[(((size_t)b * Mz + mo) * Nz + cX) * F2z + o] : 0.0f;
          outU[(((size_t)b * Mz + rowT) * Nz + colT) * F2z + o] =
              acc[v] + ob[t] + res;
        }
      }
    }
    // next iteration's post-load barrier also orders Phase C reads vs writes
  }
}

// ---------------------------------------------------------------------------
// Fused head: act = left + right(m>0); 128->64 relu; 64->64 relu; 64->256.
// One block per 128 pixels; wave w owns row-tile mt = w; A-frags hoisted.
// ---------------------------------------------------------------------------
__global__ void __launch_bounds__(256)
head_kernel(const float* __restrict__ leftU, const float* __restrict__ rightU,
            const __bf16* __restrict__ wHead,
            const float* __restrict__ oc1b, const float* __restrict__ oc2b,
            const float* __restrict__ headb, float* __restrict__ out) {
  extern __shared__ char smem[];
  __bf16* sW1 = (__bf16*)smem;      // 128x64  (8192 el)
  __bf16* sW2 = sW1 + 8192;         // 64x64   (4096 el)
  __bf16* sW3 = sW2 + 4096;         // 64x256  (16384 el)
  __bf16* sA  = sW3 + 16384;        // 128x128 act (later reused 128x64)
  __bf16* sB  = sA + 16384;         // 128x64

  const int tid  = threadIdx.x;
  const int wave = tid >> 5;
  const int lane = tid & 31;
  const int hf   = lane >> 4;
  const int l15  = lane & 15;
  const int m    = wave * 16 + l15;       // A row (mt = wave)
  const size_t p0 = (size_t)blockIdx.x * 128;

  { // weights -> LDS (28672 bf16 = 3584 uint4, contiguous in ws)
    const uint4* src = (const uint4*)wHead;
    uint4* dst = (uint4*)smem;
    for (int i = tid; i < 3584; i += 256) dst[i] = src[i];
  }
  for (int i = tid; i < 128 * 128; i += 256) {  // combine directions -> bf16
    int r = i >> 7, cch = i & 127;
    size_t p = p0 + r;
    int mm = (int)((p >> 6) & 63);
    float v = leftU[p * F2z + cch];
    if (mm > 0) v += rightU[p * F2z + cch];     // row 0 of right dir is zero
    sA[i] = (__bf16)v;
  }
  __syncthreads();

  // GEMM1: (128x128)@(128x64) + bias, relu -> sB
  {
    AF af[4];
    #pragma unroll
    for (int kt = 0; kt < 4; ++kt) {
      int k0 = kt * 32 + hf * 8;
      af[kt].q[0] = *(const uint4*)(sA + m * 128 + k0);
      af[kt].q[1] = *(const uint4*)(sA + m * 128 + k0 + 16);
    }
    #pragma unroll
    for (int nt = 0; nt < 4; ++nt) {
      v8f acc = {};
      #pragma unroll
      for (int kt = 0; kt < 4; ++kt)
        acc = wmma_bf16(af[kt].v,
            *(const v16bf*)(sW1 + (kt * 4 + nt) * 512 + lane * 16), acc);
      #pragma unroll
      for (int v = 0; v < 8; ++v) {
        int r = wave * 16 + hf * 8 + v, o = nt * 16 + l15;
        sB[r * 64 + o] = (__bf16)fmaxf(acc[v] + oc1b[o], 0.0f);
      }
    }
  }
  __syncthreads();

  // GEMM2: (128x64)@(64x64) + bias, relu -> sA (reused, stride 64)
  {
    AF af[2];
    #pragma unroll
    for (int kt = 0; kt < 2; ++kt) {
      int k0 = kt * 32 + hf * 8;
      af[kt].q[0] = *(const uint4*)(sB + m * 64 + k0);
      af[kt].q[1] = *(const uint4*)(sB + m * 64 + k0 + 16);
    }
    #pragma unroll
    for (int nt = 0; nt < 4; ++nt) {
      v8f acc = {};
      #pragma unroll
      for (int kt = 0; kt < 2; ++kt)
        acc = wmma_bf16(af[kt].v,
            *(const v16bf*)(sW2 + (kt * 4 + nt) * 512 + lane * 16), acc);
      #pragma unroll
      for (int v = 0; v < 8; ++v) {
        int r = wave * 16 + hf * 8 + v, o = nt * 16 + l15;
        sA[r * 64 + o] = (__bf16)fmaxf(acc[v] + oc2b[o], 0.0f);
      }
    }
  }
  __syncthreads();

  // GEMM3: (128x64)@(64x256) + head_b -> global logits
  {
    AF af[2];
    #pragma unroll
    for (int kt = 0; kt < 2; ++kt) {
      int k0 = kt * 32 + hf * 8;
      af[kt].q[0] = *(const uint4*)(sA + m * 64 + k0);
      af[kt].q[1] = *(const uint4*)(sA + m * 64 + k0 + 16);
    }
    #pragma unroll 1
    for (int nt = 0; nt < 16; ++nt) {
      v8f acc = {};
      #pragma unroll
      for (int kt = 0; kt < 2; ++kt)
        acc = wmma_bf16(af[kt].v,
            *(const v16bf*)(sW3 + (kt * 16 + nt) * 512 + lane * 16), acc);
      #pragma unroll
      for (int v = 0; v < 8; ++v) {
        int r = wave * 16 + hf * 8 + v, o = nt * 16 + l15;
        out[(p0 + r) * 256 + o] = acc[v] + headb[o];
      }
    }
  }
}

// ---------------------------------------------------------------------------
extern "C" void kernel_launch(void* const* d_in, const int* in_sizes, int n_in,
                              void* d_out, int out_size, void* d_ws, size_t ws_size,
                              hipStream_t stream) {
  // setup_inputs order: im, in_conv_w, in_conv_b,
  //   left{is_w,is_b,ss_w,ss_b,out_w,out_b,h0},
  //   right{...}, oc1_w, oc1_b, oc2_w, oc2_b, head_w, head_b
  const float* im        = (const float*)d_in[0];
  const float* in_conv_w = (const float*)d_in[1];
  const float* in_conv_b = (const float*)d_in[2];
  const float* L_is_w  = (const float*)d_in[3];
  const float* L_is_b  = (const float*)d_in[4];
  const float* L_ss_w  = (const float*)d_in[5];
  const float* L_ss_b  = (const float*)d_in[6];
  const float* L_out_w = (const float*)d_in[7];
  const float* L_out_b = (const float*)d_in[8];
  const float* L_h0    = (const float*)d_in[9];
  const float* R_is_w  = (const float*)d_in[10];
  const float* R_is_b  = (const float*)d_in[11];
  const float* R_ss_w  = (const float*)d_in[12];
  const float* R_ss_b  = (const float*)d_in[13];
  const float* R_out_w = (const float*)d_in[14];
  const float* R_out_b = (const float*)d_in[15];
  const float* R_h0    = (const float*)d_in[16];
  const float* oc1_w  = (const float*)d_in[17];
  const float* oc1_b  = (const float*)d_in[18];
  const float* oc2_w  = (const float*)d_in[19];
  const float* oc2_b  = (const float*)d_in[20];
  const float* head_w = (const float*)d_in[21];
  const float* head_b = (const float*)d_in[22];

  // workspace carve (~236 MB)
  char* ws = (char*)d_ws;
  float*  xf32   = (float*)ws;   ws += (size_t)Bz * Mz * Nz * F2z * 4;  // 67 MB
  __bf16* xbf    = (__bf16*)ws;  ws += (size_t)Bz * Mz * Nz * F2z * 2;  // 33.5 MB
  float*  leftU  = (float*)ws;   ws += (size_t)Bz * Mz * Nz * F2z * 4;
  float*  rightU = (float*)ws;   ws += (size_t)Bz * Mz * Nz * F2z * 4;
  __bf16* wL     = (__bf16*)ws;  ws += 73728 * 2;  // wis|wss|wout (left)
  __bf16* wR     = (__bf16*)ws;  ws += 73728 * 2;  // wis|wss|wout (right)
  __bf16* wHead  = (__bf16*)ws;  ws += 28672 * 2;  // woc1|woc2|whead

  // weight prep (swizzle into WMMA-B fragment order, fp32 -> bf16)
  swizzle_kernel<<<128, 256, 0, stream>>>(L_is_w,  wL,          128, 256, 1);
  swizzle_kernel<<<128, 256, 0, stream>>>(L_ss_w,  wL + 32768,  128, 256, 0);
  swizzle_kernel<<< 32, 256, 0, stream>>>(L_out_w, wL + 65536,   64, 128, 0);
  swizzle_kernel<<<128, 256, 0, stream>>>(R_is_w,  wR,          128, 256, 1);
  swizzle_kernel<<<128, 256, 0, stream>>>(R_ss_w,  wR + 32768,  128, 256, 0);
  swizzle_kernel<<< 32, 256, 0, stream>>>(R_out_w, wR + 65536,   64, 128, 0);
  swizzle_kernel<<< 32, 256, 0, stream>>>(oc1_w,   wHead,        128, 64, 0);
  swizzle_kernel<<< 16, 256, 0, stream>>>(oc2_w,   wHead + 8192,  64, 64, 0);
  swizzle_kernel<<< 64, 256, 0, stream>>>(head_w,  wHead + 12288, 64, 256, 0);

  // masked 7x7 causal conv
  inconv_kernel<<<Bz * Mz * Nz, 128, 0, stream>>>(im, in_conv_w, in_conv_b,
                                                  xf32, xbf);

  // bidirectional diagonal LSTM scan: 64 WGs (32 left + 32 right), 8 waves ea.
  size_t scanLds = (32768 + 32768 + 8192 + 4096) * 2;  // 155,648 B < 320 KB/WGP
  scan_kernel<<<64, 256, scanLds, stream>>>(xbf, xf32, wL, wR,
                                            L_is_b, L_ss_b, L_out_b, L_h0,
                                            R_is_b, R_ss_b, R_out_b, R_h0,
                                            leftU, rightU);

  // fused MLP head -> logits
  size_t headLds = (8192 + 4096 + 16384 + 16384 + 8192) * 2;  // 106,496 B
  head_kernel<<<(Bz * Mz * Nz) / 128, 256, headLds, stream>>>(
      leftU, rightU, wHead, oc1_b, oc2_b, head_b, (float*)d_out);
}
